// SparseSelfAttention_50251117363622
// MI455X (gfx1250) — matrix operand order repaired and checked by reading, not compiled, not run on hardware
//
#include <hip/hip_runtime.h>

typedef __attribute__((ext_vector_type(16))) _Float16 v16h;
typedef __attribute__((ext_vector_type(8)))  _Float16 v8h;
typedef __attribute__((ext_vector_type(8)))  float    v8f;
typedef __attribute__((ext_vector_type(4)))  float    v4f;

#define T_SEQ  4096
#define BATCH  2
#define NHEAD  16
#define HDIM   64
#define EMB    (NHEAD*HDIM)      // 1024
#define ROWSTR (BATCH*EMB)       // 2048 floats per t-row
#define CHUNK  128
#define LCTX   1024
#define WIN    (LCTX+CHUNK)      // 1152 = 18 * 64
#define KT     64
// 1/sqrt(64) * log2(e): softmax runs in base-2 domain (native v_exp_f32)
#define QK_SCALE_LOG2E 0.180336880260131f

// single-instruction max (avoids clang's NaN-canonicalize self-max)
__device__ __forceinline__ float vmax(float a, float b) {
    float r;
    asm("v_max_num_f32 %0, %1, %2" : "=v"(r) : "v"(a), "v"(b));
    return r;
}

// ---- 16-lane butterfly max via DPP (pure VALU) ----
template <int CTRL>
__device__ __forceinline__ float dpp_xfer(float x) {
    return __int_as_float(__builtin_amdgcn_update_dpp(
        0, __float_as_int(x), CTRL, 0xF, 0xF, true));
}
__device__ __forceinline__ float row_max16(float x) {
    x = vmax(x, dpp_xfer<0xB1>(x));   // quad_perm [1,0,3,2]  (xor 1)
    x = vmax(x, dpp_xfer<0x4E>(x));   // quad_perm [2,3,0,1]  (xor 2)
    x = vmax(x, dpp_xfer<0x141>(x));  // row_half_mirror      (xor 7)
    x = vmax(x, dpp_xfer<0x140>(x));  // row_mirror           (xor 15)
    return x;
}

__global__ __launch_bounds__(256)
void sparse_attn_kernel(const float* __restrict__ q,
                        const float* __restrict__ k,
                        const float* __restrict__ v,
                        float* __restrict__ out)
{
    __shared__ alignas(16) _Float16 Qs[CHUNK*HDIM];   // 128x64 f16, pre-scaled
    __shared__ alignas(16) _Float16 Ks[KT*HDIM];      // 64x64  key-major
    __shared__ alignas(16) _Float16 Vts[HDIM*KT];     // 64x64  dim-major (V^T)
    __shared__ alignas(16) _Float16 Ps[8*16*KT];      // per-wave P scratch (16x64)

    const int tid  = threadIdx.x;
    const int wave = tid >> 5;
    const int lane = tid & 31;
    const int lhi  = lane >> 4;
    const int llo  = lane & 15;

    const int ch = blockIdx.x;            // chunk 0..31
    const int bh = blockIdx.y;            // batch*head 0..31
    const int b  = bh >> 4;
    const int h  = bh & 15;
    const long base_e = (long)b*EMB + h*HDIM;

    const int srow = tid >> 3;            // 0..31 : staging row (per 32-row half)
    const int scol = (tid & 7) * 8;       // 0..56 : staging col base (8 floats)

    // ---- stage Q (pre-scaled into log2 domain) into LDS as f16 ----
    #pragma unroll
    for (int it = 0; it < 4; ++it) {
        const int row = it*32 + srow;
        const long off = (long)(ch*CHUNK + row)*ROWSTR + base_e + scol;
        v4f q0 = *(const v4f*)&q[off];
        v4f q1 = *(const v4f*)&q[off + 4];
        v8h hq;
        #pragma unroll
        for (int i = 0; i < 4; ++i) {
            hq[i]   = (_Float16)(q0[i] * QK_SCALE_LOG2E);
            hq[4+i] = (_Float16)(q1[i] * QK_SCALE_LOG2E);
        }
        *(v8h*)&Qs[row*HDIM + scol] = hq;
    }
    __syncthreads();

    // ---- per-wave Q A-fragments (rows m0..m0+15, K = 64 = 2 steps) ----
    const int m0 = wave * 16;
    v16h aq0, aq1;
    {
        const _Float16* qrow = &Qs[(m0 + llo) * HDIM];
        const int kb = lhi * 8;
        #pragma unroll
        for (int i = 0; i < 8; ++i) {
            aq0[i]   = qrow[kb + i];
            aq0[8+i] = qrow[16 + kb + i];
            aq1[i]   = qrow[32 + kb + i];
            aq1[8+i] = qrow[48 + kb + i];
        }
    }

    // constant ones-column B fragment (column n==0 of a 32x16 tile):
    // P x ones accumulates per-row softmax denominators on the WMMA pipe.
    v16h onesb;
    {
        const _Float16 ov = (llo == 0) ? (_Float16)1.0f : (_Float16)0.0f;
        #pragma unroll
        for (int i = 0; i < 16; ++i) onesb[i] = ov;
    }

    float mrow[8];
    v8f acc[4], acc_l;
    #pragma unroll
    for (int r = 0; r < 8; ++r) { mrow[r] = -30000.f; acc_l[r] = 0.f; }
    #pragma unroll
    for (int nt = 0; nt < 4; ++nt)
        #pragma unroll
        for (int r = 0; r < 8; ++r) acc[nt][r] = 0.f;

    const int jstart = (ch < 8) ? (LCTX - ch*CHUNK) : 0;  // multiple of 128
    const int kv_t0  = ch*CHUNK - LCTX;

    // ---- software-pipelined K/V staging registers (2 x 32-row halves) ----
    v4f kr[2][2], vr[2][2];
    auto fetch_kv = [&](int j0) {
        #pragma unroll
        for (int it = 0; it < 2; ++it) {
            const int tk = kv_t0 + j0 + it*32 + srow;
            if (tk >= 0) {
                const long off = (long)tk*ROWSTR + base_e + scol;
                kr[it][0] = *(const v4f*)&k[off];
                kr[it][1] = *(const v4f*)&k[off + 4];
                vr[it][0] = *(const v4f*)&v[off];
                vr[it][1] = *(const v4f*)&v[off + 4];
            } else {
                #pragma unroll
                for (int i = 0; i < 4; ++i) {
                    kr[it][0][i]=0.f; kr[it][1][i]=0.f;
                    vr[it][0][i]=0.f; vr[it][1][i]=0.f;
                }
            }
        }
    };

    fetch_kv(jstart);

    for (int j0 = jstart; j0 < WIN; j0 += KT) {
        // ---- stage current tile regs -> LDS ----
        #pragma unroll
        for (int it = 0; it < 2; ++it) {
            const int row = it*32 + srow;
            v8h hk, hv;
            #pragma unroll
            for (int i = 0; i < 4; ++i) {
                hk[i]   = (_Float16)kr[it][0][i];  hk[4+i] = (_Float16)kr[it][1][i];
                hv[i]   = (_Float16)vr[it][0][i];  hv[4+i] = (_Float16)vr[it][1][i];
            }
            *(v8h*)&Ks[row*HDIM + scol] = hk;
            #pragma unroll
            for (int i = 0; i < 8; ++i) Vts[(scol + i)*KT + row] = hv[i];
        }
        __syncthreads();

        // fetch next tile (latency hidden behind this tile's compute)
        if (j0 + KT < WIN) fetch_kv(j0 + KT);
        // warm the tile after that (64 rows x 256B each)
        if (j0 + 2*KT < WIN) {
            const int tk = kv_t0 + j0 + 2*KT + (tid >> 2);
            if (tk >= 0) {
                const long off = (long)tk*ROWSTR + base_e + (tid & 3)*16;
                __builtin_prefetch(&k[off], 0, 3);
                __builtin_prefetch(&v[off], 0, 3);
            }
        }

        // wave-uniform sparsity: a 64-key tile never straddles a 128-block,
        // and qb == m0 for the whole stripe.  Skipped tiles contribute 0.
        const bool wave_active = (((j0 >> 7) << 7) >= m0) && (j0 - LCTX <= m0 + 15);
        if (wave_active) {
            // ---- S = Q K^T : four 16-key subtiles x 2 chained WMMAs ----
            v8f s[4];
            #pragma unroll
            for (int f = 0; f < 4; ++f) {
                v16h bk0, bk1;
                const _Float16* krow = &Ks[(f*16 + llo) * HDIM + lhi*8];
                #pragma unroll
                for (int i = 0; i < 8; ++i) { bk0[i] = krow[i]; bk0[8+i] = krow[16 + i]; bk1[i] = krow[32 + i]; bk1[8+i] = krow[48 + i]; }
                v8f cz;
                #pragma unroll
                for (int r = 0; r < 8; ++r) cz[r] = 0.f;
                cz = __builtin_amdgcn_wmma_f32_16x16x32_f16(false, aq0, false, bk0, (short)0, cz, false, false);
                cz = __builtin_amdgcn_wmma_f32_16x16x32_f16(false, aq1, false, bk1, (short)0, cz, false, false);
                s[f] = cz;
            }

            // elementwise causal mask only needed on diagonal tiles
            const bool needs_mask = (j0 + KT - 1 - LCTX) > m0;
            _Float16* Pw = &Ps[wave * 16 * KT];

            #pragma unroll
            for (int r = 0; r < 8; ++r) {
                float t[4];
                #pragma unroll
                for (int f = 0; f < 4; ++f) t[f] = s[f][r];
                if (needs_mask) {
                    const int iloc = m0 + lhi*8 + r;
                    #pragma unroll
                    for (int f = 0; f < 4; ++f)
                        t[f] = (j0 + f*16 + llo - LCTX <= iloc) ? t[f] : -1e30f;
                }
                const float tm = row_max16(vmax(vmax(t[0], t[1]), vmax(t[2], t[3])));
                const float mn = vmax(mrow[r], tm);
                // exp2 + immediate f16 store to P scratch (no live p array)
                #pragma unroll
                for (int f = 0; f < 4; ++f)
                    Pw[(lhi*8 + r) * KT + f*16 + llo] =
                        (_Float16)__builtin_amdgcn_exp2f(t[f] - mn);
                const float sc = __builtin_amdgcn_exp2f(mrow[r] - mn);
                mrow[r] = mn;
                acc_l[r] *= sc;
                #pragma unroll
                for (int nt = 0; nt < 4; ++nt) acc[nt][r] *= sc;
            }

            // ---- A-fragments of P (16 x 64 = 2 K-steps) ----
            v16h pa0, pa1;
            {
                const _Float16* prow = &Pw[llo * KT];
                const int kb = lhi * 8;
                #pragma unroll
                for (int i = 0; i < 8; ++i) {
                    pa0[i]   = prow[kb + i];
                    pa0[8+i] = prow[16 + kb + i];
                    pa1[i]   = prow[32 + kb + i];
                    pa1[8+i] = prow[48 + kb + i];
                }
            }

            // ---- O += P * V : 4 n-tiles x 2 K-steps + 2 denominator WMMAs ----
            #pragma unroll
            for (int nt = 0; nt < 4; ++nt) {
                v16h bv0, bv1;
                const _Float16* vrow = &Vts[(nt*16 + llo) * KT + lhi*8];
                #pragma unroll
                for (int i = 0; i < 8; ++i) { bv0[i] = vrow[i]; bv0[8+i] = vrow[16 + i]; bv1[i] = vrow[32 + i]; bv1[8+i] = vrow[48 + i]; }
                acc[nt] = __builtin_amdgcn_wmma_f32_16x16x32_f16(false, pa0, false, bv0, (short)0, acc[nt], false, false);
                acc[nt] = __builtin_amdgcn_wmma_f32_16x16x32_f16(false, pa1, false, bv1, (short)0, acc[nt], false, false);
            }
            acc_l = __builtin_amdgcn_wmma_f32_16x16x32_f16(false, pa0, false, onesb, (short)0, acc_l, false, false);
            acc_l = __builtin_amdgcn_wmma_f32_16x16x32_f16(false, pa1, false, onesb, (short)0, acc_l, false, false);
        }
        __syncthreads();
    }

    // ---- epilogue: denominators sit in column 0 (lanes 0/16); broadcast,
    //      normalize and store f32 output ----
    #pragma unroll
    for (int r = 0; r < 8; ++r) {
        const float lsum = __shfl(acc_l[r], lane & 16);
        const float inv  = 1.0f / lsum;
        const int t = ch*CHUNK + m0 + lhi*8 + r;
        const long orow = (long)t*ROWSTR + base_e;
        #pragma unroll
        for (int nt = 0; nt < 4; ++nt)
            out[orow + nt*16 + llo] = acc[nt][r] * inv;
    }
}

extern "C" void kernel_launch(void* const* d_in, const int* in_sizes, int n_in,
                              void* d_out, int out_size, void* d_ws, size_t ws_size,
                              hipStream_t stream) {
    const float* q = (const float*)d_in[0];
    const float* k = (const float*)d_in[1];
    const float* v = (const float*)d_in[2];
    float* out = (float*)d_out;
    (void)in_sizes; (void)n_in; (void)out_size; (void)d_ws; (void)ws_size;
    dim3 grid(T_SEQ / CHUNK, BATCH * NHEAD);
    sparse_attn_kernel<<<grid, dim3(256), 0, stream>>>(q, k, v, out);
}
